// Decoder_18365280158321
// MI455X (gfx1250) — matrix operand order repaired and checked by reading, not compiled, not run on hardware
//
#include <hip/hip_runtime.h>
#include <math.h>

#define L_  2
#define B_  256
#define H_  1024
#define O_  256
#define T_  128
#define G4  (4 * H_)   // 4096

typedef __bf16 bf16;
typedef __attribute__((ext_vector_type(16))) __bf16 v16bf;
typedef __attribute__((ext_vector_type(8)))  float  v8f;
typedef __attribute__((ext_vector_type(4)))  unsigned int v4u;

union ABu { v16bf v; v4u u[2]; };

__device__ __forceinline__ float sigf(float x) {
    return 1.0f / (1.0f + __expf(-x));
}
__device__ __forceinline__ float tanh_fast(float x) {
    x = fminf(fmaxf(x, -15.0f), 15.0f);
    float e = __expf(-2.0f * x);
    return (1.0f - e) / (1.0f + e);
}

// ---------------- one-time prep kernels ----------------

// src: (N x K) f32 row-major  ->  dst: (K x N) bf16 row-major (i.e. src^T)
__global__ void k_transpose_bf16(const float* __restrict__ src,
                                 bf16* __restrict__ dst, int N, int K) {
    int idx = blockIdx.x * blockDim.x + threadIdx.x;
    if (idx >= N * K) return;
    int k = idx / N, n = idx % N;
    dst[idx] = (bf16)src[n * K + k];
}

// init h (bf16 ping buffers), c (f32 ws), combined layer-1 bias
__global__ void k_init_state(const float* __restrict__ h0,
                             const float* __restrict__ c0,
                             const float* __restrict__ b_ih,
                             const float* __restrict__ b_hh,
                             bf16* __restrict__ h0a, bf16* __restrict__ h1a,
                             float* __restrict__ c0w, float* __restrict__ c1w,
                             float* __restrict__ bias1) {
    int i = blockIdx.x * blockDim.x + threadIdx.x;
    if (i < B_ * H_) {
        h0a[i] = (bf16)h0[i];
        h1a[i] = (bf16)h0[B_ * H_ + i];
        c0w[i] = c0[i];
        c1w[i] = c0[B_ * H_ + i];
    }
    if (i < G4) bias1[i] = b_ih[G4 + i] + b_hh[G4 + i];
}

// xg0[b][j] = x[b] . W_ih0[j] + b_ih0[j] + b_hh0[j]   (one-time; x constant over t)
__global__ void k_xgates0(const float* __restrict__ x,
                          const float* __restrict__ Wih0,
                          const float* __restrict__ bih0,
                          const float* __restrict__ bhh0,
                          float* __restrict__ xg0) {
    int idx = blockIdx.x * blockDim.x + threadIdx.x;
    if (idx >= B_ * G4) return;
    int b = idx / G4, j = idx % G4;
    const float* xr = x + b * H_;
    const float* wr = Wih0 + j * H_;
    float s = bih0[j] + bhh0[j];
    for (int k = 0; k < H_; ++k) s = fmaf(xr[k], wr[k], s);
    xg0[idx] = s;
}

// ---------------- fused LSTM layer: 4-gate WMMA GEMM + pointwise ----------------
// Block = 8 waves covering 8 m-tiles x 1 n-tile; the 4-gate B tile (4KB per
// k-block) is staged ONE ITERATION AHEAD into a double-buffered LDS tile and
// shared by all 8 waves (8x less weight traffic from L2). Iteration i:
//   - batched ds_loads of buf(i&1)  (staged during iter i-1)
//   - ds_store of buf((i+1)&1) from a global-prefetched register (chunk i+1)
//   - global prefetch of chunk i+2 into the register
//   - A fragment for iter i+1 prefetched from global into registers
//   - 4 WMMAs (gates i,f,g,o) on registers
//   - one barrier at end of iteration
// LAYER1=false: K=H,  gate addend = xg0 (B x 4H, precomputed x-term + biases)
// LAYER1=true : K=2H, A = [A0 ; A1] (h0_new ; h1_old), addend = bias1 (4H)
template <bool LAYER1>
__global__ void __launch_bounds__(256)
k_lstm_layer(const bf16* __restrict__ A0, const bf16* __restrict__ A1,
             const bf16* __restrict__ WT,
             const float* __restrict__ addend2d,   // B x 4H (layer 0)
             const float* __restrict__ bias1,      // 4H (layer 1)
             float* __restrict__ c_io,             // B x H f32, in/out
             bf16* __restrict__ hout) {            // B x H bf16
    constexpr int K   = LAYER1 ? 2 * H_ : H_;
    constexpr int NKB = K >> 5;                    // k-blocks of 32
    __shared__ bf16 Bs[2][4 * 32 * 16];            // [buf][gate*512 + k*16 + n]

    const int tid  = threadIdx.x;
    const int lane = tid & 31;
    const int w    = tid >> 5;                     // wave 0..7
    const int NT   = H_ / 16;                      // 64 n-tiles
    const int nt   = blockIdx.x % NT;
    const int mtg  = blockIdx.x / NT;              // m-tile group 0..1
    const int mt   = mtg * 8 + w;
    const int mrow = lane & 15;                    // A-tile row for this lane
    const int k0   = (lane >> 4) * 8;              // A-tile k sub-offset

    // staging role: each thread moves one 16B chunk of the 4KB B tile
    const int sg = tid >> 6;                       // gate 0..3
    const int sr = (tid >> 1) & 31;                // k-row 0..31
    const int sh = tid & 1;                        // 16B half 0..1
    const bf16* wbase = WT + (size_t)sr * G4 + nt * 16 + sg * H_ + sh * 8;
    bf16* sdst0 = &Bs[0][sg * 512 + sr * 16 + sh * 8];
    bf16* sdst1 = &Bs[1][sg * 512 + sr * 16 + sh * 8];

    const bf16* Arow0 = A0 + (size_t)(mt * 16 + mrow) * H_ + k0;
    const bf16* Arow1 = LAYER1 ? (A1 + (size_t)(mt * 16 + mrow) * H_ + k0) : Arow0;

    v8f acc[4];
    #pragma unroll
    for (int g = 0; g < 4; ++g)
        #pragma unroll
        for (int r = 0; r < 8; ++r) acc[g][r] = 0.0f;

    // ---- prologue: stage chunk 0 into buf0, prefetch chunk 1 + A(0) ----
    v4u breg = *(const v4u*)(wbase);
    *(v4u*)sdst0 = breg;
    breg = *(const v4u*)(wbase + (size_t)32 * G4);
    ABu a_cur;
    a_cur.u[0] = *(const v4u*)(Arow0);
    a_cur.u[1] = *(const v4u*)(Arow0 + 16);
    __syncthreads();

    for (int i = 0; i < NKB; ++i) {
        const int buf = i & 1;
        // batched LDS fragment loads for this iteration's B tile
        v16bf bv0, bv1, bv2, bv3;
        #pragma unroll
        for (int s = 0; s < 16; ++s) {
            const int kk = (s < 8) ? (k0 + s) : (16 + k0 + (s - 8));
            const int bidx = kk * 16 + (lane & 15);
            bv0[s] = Bs[buf][0 * 512 + bidx];
            bv1[s] = Bs[buf][1 * 512 + bidx];
            bv2[s] = Bs[buf][2 * 512 + bidx];
            bv3[s] = Bs[buf][3 * 512 + bidx];
        }
        // stage chunk i+1 into the other buffer; prefetch chunk i+2
        if (i + 1 < NKB) {
            *(v4u*)(buf ? sdst0 : sdst1) = breg;
            if (i + 2 < NKB)
                breg = *(const v4u*)(wbase + (size_t)((i + 2) << 5) * G4);
        }
        // prefetch A fragment for iteration i+1
        ABu a_next = a_cur;
        if (i + 1 < NKB) {
            const int kb = (i + 1) << 5;
            const bf16* ap;
            if (!LAYER1)  ap = Arow0 + kb;
            else          ap = (kb < H_) ? (Arow0 + kb) : (Arow1 + (kb - H_));
            a_next.u[0] = *(const v4u*)(ap);
            a_next.u[1] = *(const v4u*)(ap + 16);
        }
        // 4-gate WMMA on registers
        acc[0] = __builtin_amdgcn_wmma_f32_16x16x32_bf16(
            false, a_cur.v, false, bv0, (short)0, acc[0], false, false);
        acc[1] = __builtin_amdgcn_wmma_f32_16x16x32_bf16(
            false, a_cur.v, false, bv1, (short)0, acc[1], false, false);
        acc[2] = __builtin_amdgcn_wmma_f32_16x16x32_bf16(
            false, a_cur.v, false, bv2, (short)0, acc[2], false, false);
        acc[3] = __builtin_amdgcn_wmma_f32_16x16x32_bf16(
            false, a_cur.v, false, bv3, (short)0, acc[3], false, false);
        a_cur = a_next;
        __syncthreads();
    }

    // ---- pointwise LSTM epilogue (lane-local: all gates share C/D layout) ----
    const int n     = lane & 15;
    const int rbase = (lane >> 4) * 8;
    const int col   = nt * 16 + n;
    float bi = 0.f, bf = 0.f, bg = 0.f, bo = 0.f;
    if (LAYER1) {
        bi = bias1[col]; bf = bias1[H_ + col];
        bg = bias1[2 * H_ + col]; bo = bias1[3 * H_ + col];
    }
    #pragma unroll
    for (int r = 0; r < 8; ++r) {
        int row = mt * 16 + rbase + r;
        float gi = acc[0][r] + bi, gf = acc[1][r] + bf;
        float gg = acc[2][r] + bg, go = acc[3][r] + bo;
        if (!LAYER1) {
            const float* ad = addend2d + (size_t)row * G4 + col;
            gi += ad[0]; gf += ad[H_]; gg += ad[2 * H_]; go += ad[3 * H_];
        }
        float iv = sigf(gi), fv = sigf(gf), gv = tanh_fast(gg), ov = sigf(go);
        float cn = fmaf(fv, c_io[row * H_ + col], iv * gv);
        c_io[row * H_ + col] = cn;
        hout[row * H_ + col] = (bf16)(ov * tanh_fast(cn));
    }
}

// ---------------- FC GEMM (WMMA): pred[b][t][o] = h1 . fc_w^T + fc_b ----------------
__global__ void __launch_bounds__(256)
k_fc(const bf16* __restrict__ A,        // B x H bf16 (h1_new)
     const bf16* __restrict__ WT,       // H x O bf16 (fc_w^T)
     const float* __restrict__ bias,    // O
     float* __restrict__ out, int t) {
    const int lane = threadIdx.x & 31;
    const int wid  = blockIdx.x * (blockDim.x >> 5) + (threadIdx.x >> 5);
    const int NT   = O_ / 16;            // 16
    const int mt   = wid / NT;
    const int nt   = wid % NT;
    const int mrow = lane & 15;
    const int k0   = (lane >> 4) * 8;

    v8f acc;
    #pragma unroll
    for (int r = 0; r < 8; ++r) acc[r] = 0.0f;

    for (int kb = 0; kb < H_; kb += 32) {
        ABu a;
        const bf16* ap = A + (size_t)(mt * 16 + mrow) * H_ + kb + k0;
        a.u[0] = *(const v4u*)(ap);
        a.u[1] = *(const v4u*)(ap + 16);
        v16bf bv;
        #pragma unroll
        for (int s = 0; s < 16; ++s) {
            const int kk = (s < 8) ? (k0 + s) : (16 + k0 + (s - 8));
            bv[s] = WT[(size_t)(kb + kk) * O_ + nt * 16 + mrow];
        }
        acc = __builtin_amdgcn_wmma_f32_16x16x32_bf16(
            false, a.v, false, bv, (short)0, acc, false, false);
    }

    const int n = lane & 15;
    const int rbase = (lane >> 4) * 8;
    const int col = nt * 16 + n;
    float bb = bias[col];
    #pragma unroll
    for (int r = 0; r < 8; ++r) {
        int row = mt * 16 + rbase + r;
        out[(size_t)row * (T_ * O_) + t * O_ + col] = acc[r] + bb;  // (B, T, O)
    }
}

// ---------------- log_softmax (first O-1) + sigmoid (last), in place ----------------
__global__ void __launch_bounds__(256)
k_post(float* __restrict__ out) {
    __shared__ float red[O_];
    const int base = blockIdx.x * O_;
    const int j = threadIdx.x;
    float v = out[base + j];
    red[j] = (j < O_ - 1) ? v : -3.4e38f;
    __syncthreads();
    for (int s = 128; s > 0; s >>= 1) {
        if (j < s) red[j] = fmaxf(red[j], red[j + s]);
        __syncthreads();
    }
    float mx = red[0];
    __syncthreads();
    red[j] = (j < O_ - 1) ? __expf(v - mx) : 0.0f;
    __syncthreads();
    for (int s = 128; s > 0; s >>= 1) {
        if (j < s) red[j] += red[j + s];
        __syncthreads();
    }
    float lse = __logf(red[0]) + mx;
    out[base + j] = (j < O_ - 1) ? (v - lse) : sigf(v);
}

// ---------------- host ----------------

extern "C" void kernel_launch(void* const* d_in, const int* in_sizes, int n_in,
                              void* d_out, int out_size, void* d_ws, size_t ws_size,
                              hipStream_t stream) {
    const float* h0   = (const float*)d_in[0];   // (2,B,H)
    const float* c0   = (const float*)d_in[1];   // (2,B,H)
    const float* x    = (const float*)d_in[2];   // (B,1,H)
    const float* W_ih = (const float*)d_in[3];   // (2,4H,H)
    const float* W_hh = (const float*)d_in[4];   // (2,4H,H)
    const float* b_ih = (const float*)d_in[5];   // (2,4H)
    const float* b_hh = (const float*)d_in[6];   // (2,4H)
    const float* fc_w = (const float*)d_in[7];   // (O,H)
    const float* fc_b = (const float*)d_in[8];   // (O)
    float* out = (float*)d_out;                  // (B,T,O)

    char* p = (char*)d_ws;
    auto take = [&](size_t bytes) { char* r = p; p += (bytes + 255) & ~size_t(255); return r; };
    bf16*  WT_hh0  = (bf16*)take((size_t)H_ * G4 * 2);        // H x 4H
    bf16*  WT_cat1 = (bf16*)take((size_t)2 * H_ * G4 * 2);    // 2H x 4H  ([W_ih1;W_hh1]^T)
    bf16*  WT_fc   = (bf16*)take((size_t)H_ * O_ * 2);        // H x O
    float* xg0     = (float*)take((size_t)B_ * G4 * 4);       // B x 4H
    float* bias1   = (float*)take((size_t)G4 * 4);
    float* c0w     = (float*)take((size_t)B_ * H_ * 4);
    float* c1w     = (float*)take((size_t)B_ * H_ * 4);
    bf16*  h0a     = (bf16*)take((size_t)B_ * H_ * 2);
    bf16*  h0b     = (bf16*)take((size_t)B_ * H_ * 2);
    bf16*  h1a     = (bf16*)take((size_t)B_ * H_ * 2);
    bf16*  h1b     = (bf16*)take((size_t)B_ * H_ * 2);

    const int TPB = 256;
    const int nTr = G4 * H_;
    // W_hh[0]^T
    k_transpose_bf16<<<(nTr + TPB - 1) / TPB, TPB, 0, stream>>>(W_hh, WT_hh0, G4, H_);
    // [W_ih[1]; W_hh[1]]^T stacked along K
    k_transpose_bf16<<<(nTr + TPB - 1) / TPB, TPB, 0, stream>>>(W_ih + (size_t)G4 * H_, WT_cat1, G4, H_);
    k_transpose_bf16<<<(nTr + TPB - 1) / TPB, TPB, 0, stream>>>(W_hh + (size_t)G4 * H_, WT_cat1 + (size_t)H_ * G4, G4, H_);
    // fc_w^T
    k_transpose_bf16<<<(O_ * H_ + TPB - 1) / TPB, TPB, 0, stream>>>(fc_w, WT_fc, O_, H_);

    k_init_state<<<(B_ * H_ + TPB - 1) / TPB, TPB, 0, stream>>>(h0, c0, b_ih, b_hh,
                                                                h0a, h1a, c0w, c1w, bias1);
    k_xgates0<<<(B_ * G4 + TPB - 1) / TPB, TPB, 0, stream>>>(x, W_ih, b_ih, b_hh, xg0);

    // layer kernels: 128 blocks = (16 m-tiles / 8 per block) * 64 n-tiles
    const int blocksL = (B_ / 16 / 8) * (H_ / 16);   // 128
    const int wavesF  = (B_ / 16) * (O_ / 16);       // 256
    const int blocksF = wavesF / (TPB / 32);         // 32

    bf16 *h0r = h0a, *h0w = h0b, *h1r = h1a, *h1w = h1b;
    for (int t = 0; t < T_; ++t) {
        // layer 0: gates = h0_old @ W_hh0^T + xg0 (x-term + biases, precomputed)
        k_lstm_layer<false><<<blocksL, TPB, 0, stream>>>(h0r, nullptr, WT_hh0,
                                                         xg0, nullptr, c0w, h0w);
        // layer 1: gates = [h0_new ; h1_old] @ [W_ih1;W_hh1]^T + bias1
        k_lstm_layer<true><<<blocksL, TPB, 0, stream>>>(h0w, h1r, WT_cat1,
                                                        nullptr, bias1, c1w, h1w);
        // FC: pred_t = h1_new @ fc_w^T + fc_b  -> out[:, t, :]
        k_fc<<<blocksF, TPB, 0, stream>>>(h1w, WT_fc, fc_b, out, t);

        bf16* tmp;
        tmp = h0r; h0r = h0w; h0w = tmp;
        tmp = h1r; h1r = h1w; h1w = tmp;
    }

    k_post<<<B_ * T_, TPB, 0, stream>>>(out);
}
